// GATM_14929306321145
// MI455X (gfx1250) — matrix operand order repaired and checked by reading, not compiled, not run on hardware
//
#include <hip/hip_runtime.h>
#include <hip/hip_bf16.h>

typedef _Float16 f16;
typedef __attribute__((ext_vector_type(16))) _Float16 v16h;
typedef __attribute__((ext_vector_type(8)))  _Float16 v8h;
typedef __attribute__((ext_vector_type(8)))  float    v8f;

#define NEG_SLOPE 0.2f
#define EPS_F 1e-16f

// ---------------- order-preserving float <-> uint for atomic max ----------------
__device__ __forceinline__ unsigned int fenc(float f) {
    unsigned int u = __float_as_uint(f);
    return (u & 0x80000000u) ? ~u : (u | 0x80000000u);
}
__device__ __forceinline__ float fdec(unsigned int u) {
    return (u & 0x80000000u) ? __uint_as_float(u ^ 0x80000000u) : __uint_as_float(~u);
}

// ---------------- f32 -> f16 convert ----------------
__global__ void k_f32_to_f16(const float* __restrict__ x, f16* __restrict__ y, long long n) {
    long long i = (long long)blockIdx.x * blockDim.x + threadIdx.x;
    long long stride = (long long)gridDim.x * blockDim.x;
    for (; i < n; i += stride) y[i] = (f16)x[i];
}

// ---------------- W[k][n] f32 -> Wt[n][k] f16 ----------------
__global__ void k_transpose_to_f16(const float* __restrict__ W, f16* __restrict__ Wt,
                                   int K, int Nn) {
    int idx = blockIdx.x * blockDim.x + threadIdx.x;
    if (idx >= K * Nn) return;
    int n = idx / K, k = idx % K;
    Wt[(size_t)n * K + k] = (f16)W[(size_t)k * Nn + n];
}

// ---------------- WMMA GEMM: C[M,Nn] = A[M,128] * Bt[Nn,128]^T ----------------
// One wave per 16-row M tile; NTILES 16-col tiles across N. K = 128 (4 chunks of 32).
template <int NTILES>
__global__ __launch_bounds__(256) void k_gemm_wmma(const f16* __restrict__ A,
                                                   const f16* __restrict__ Bt,
                                                   float* __restrict__ C,
                                                   int M, int Mtiles) {
    int wave  = threadIdx.x >> 5;
    int mtile = blockIdx.x * (blockDim.x >> 5) + wave;
    if (mtile >= Mtiles) return;                 // wave-uniform: EXEC all-1s below
    int lane = threadIdx.x & 31;
    int mr   = lane & 15;                        // M row (A) / N col (B,C) within tile
    int hi   = lane >> 4;                        // lane half

    v8f acc[NTILES] = {};

    int row = mtile * 16 + mr;
    if (row >= M) row = M - 1;                   // clamp reads (stores guarded below)
    const f16* arow = A + (size_t)row * 128;

#pragma unroll
    for (int kc = 0; kc < 4; ++kc) {
        const int kb = kc * 32;
        // A fragment (ISA 16-bit A 16x32 layout): e0..7 <- K=kb+hi*8.., e8..15 <- K=kb+16+hi*8..
        v8h a0 = *(const v8h*)(arow + kb + hi * 8);
        v8h a1 = *(const v8h*)(arow + kb + 16 + hi * 8);
        v16h af;
#pragma unroll
        for (int i = 0; i < 8; ++i) { af[i] = a0[i]; af[8 + i] = a1[i]; }
#pragma unroll
        for (int t = 0; t < NTILES; ++t) {
            // B fragment (32x16): lane n=mr holds e0..7 <- K=kb+hi*8.., e8..15 <- K=kb+16+hi*8.. in Bt[n][k]
            const f16* bp = Bt + (size_t)(t * 16 + mr) * 128 + kb;
            v8h b0 = *(const v8h*)(bp + hi * 8);
            v8h b1 = *(const v8h*)(bp + 16 + hi * 8);
            v16h bf;
#pragma unroll
            for (int i = 0; i < 8; ++i) { bf[i] = b0[i]; bf[8 + i] = b1[i]; }
            acc[t] = __builtin_amdgcn_wmma_f32_16x16x32_f16(
                false, af, false, bf, (short)0, acc[t], false, false);
        }
    }

    const int Nn = NTILES * 16;
#pragma unroll
    for (int t = 0; t < NTILES; ++t) {
        float* cp = C + (size_t)(mtile * 16) * Nn + t * 16 + mr;
#pragma unroll
        for (int e = 0; e < 8; ++e) {
            int r = mtile * 16 + hi * 8 + e;     // C layout: VGPR e -> M = e + 8*half
            if (r < M) cp[(size_t)(hi * 8 + e) * Nn] = acc[t][e];
        }
    }
}

// ---------------- per-node attention scores: a_s/a_d [N,H] ----------------
__global__ void k_node_att(const float* __restrict__ h, const float* __restrict__ att_src,
                           const float* __restrict__ att_dst, float* __restrict__ a_s,
                           float* __restrict__ a_d, int Nnodes, int H, int Cc) {
    int idx = blockIdx.x * blockDim.x + threadIdx.x;
    if (idx >= Nnodes * H) return;
    int node = idx / H, hd = idx % H;
    const float* hp = h + (size_t)node * H * Cc + (size_t)hd * Cc;
    const float* as = att_src + hd * Cc;
    const float* ad = att_dst + hd * Cc;
    float s = 0.f, d = 0.f;
    for (int c = 0; c < Cc; ++c) { float v = hp[c]; s += v * as[c]; d += v * ad[c]; }
    a_s[idx] = s;
    a_d[idx] = d;
}

// ---------------- init m (encoded -inf == 0) and denom ----------------
__global__ void k_init_softmax(unsigned int* __restrict__ m, float* __restrict__ denom,
                               long long n) {
    long long i = (long long)blockIdx.x * blockDim.x + threadIdx.x;
    if (i < n) { m[i] = 0u; denom[i] = 0.f; }
}

// ---------------- init output with bias ----------------
__global__ void k_init_bias(float* __restrict__ out, const float* __restrict__ bias,
                            long long Nnodes, int F) {
    long long i = (long long)blockIdx.x * blockDim.x + threadIdx.x;
    if (i < Nnodes * (long long)F) out[i] = bias[i % F];
}

__device__ __forceinline__ void edge_nodes(const long long* __restrict__ ei, long long Eg,
                                           long long e, long long& s, long long& d) {
    if (e < Eg) { s = ei[e]; d = ei[Eg + e]; }
    else        { s = e - Eg; d = s; }         // self-loops appended
}

__device__ __forceinline__ float lrelu(float v) { return v >= 0.f ? v : NEG_SLOPE * v; }

// ---------------- pass 1: segment max over dst (thread per edge, loop heads) ----------
__global__ void k_edge_max(const long long* __restrict__ ei, long long Eg, long long Etot,
                           const float* __restrict__ a_s, const float* __restrict__ a_d,
                           unsigned int* __restrict__ m, int H) {
    long long e = (long long)blockIdx.x * blockDim.x + threadIdx.x;
    long long stride = (long long)gridDim.x * blockDim.x;
    for (; e < Etot; e += stride) {
        long long s, d; edge_nodes(ei, Eg, e, s, d);
        for (int hd = 0; hd < H; ++hd) {
            float v = lrelu(a_s[s * H + hd] + a_d[d * H + hd]);
            atomicMax(&m[d * H + hd], fenc(v));
        }
    }
}

// ---------------- pass 2: segment sum of exp(e - m) ----------------
__global__ void k_edge_expsum(const long long* __restrict__ ei, long long Eg, long long Etot,
                              const float* __restrict__ a_s, const float* __restrict__ a_d,
                              const unsigned int* __restrict__ m, float* __restrict__ denom,
                              int H) {
    long long e = (long long)blockIdx.x * blockDim.x + threadIdx.x;
    long long stride = (long long)gridDim.x * blockDim.x;
    for (; e < Etot; e += stride) {
        long long s, d; edge_nodes(ei, Eg, e, s, d);
        for (int hd = 0; hd < H; ++hd) {
            float v = lrelu(a_s[s * H + hd] + a_d[d * H + hd]);
            float mx = fdec(m[d * H + hd]);
            atomicAdd(&denom[d * H + hd], __expf(v - mx));
        }
    }
}

// ---------------- pass 3: out[dst] += h[src] * alpha (one wave per edge) ----------------
// Lanes 0..H-1 compute alpha once; channel lanes fetch it via cross-lane shuffle.
__global__ __launch_bounds__(256) void k_edge_aggregate(
    const long long* __restrict__ ei, long long Eg, long long Etot,
    const float* __restrict__ a_s, const float* __restrict__ a_d,
    const unsigned int* __restrict__ m, const float* __restrict__ denom,
    const float* __restrict__ hbuf, float* __restrict__ out, int H, int Cc) {
    const int F = H * Cc;
    long long wpb  = blockDim.x >> 5;
    long long widx = (long long)blockIdx.x * wpb + (threadIdx.x >> 5);
    int lane = threadIdx.x & 31;
    long long wstride = (long long)gridDim.x * wpb;
    for (long long e = widx; e < Etot; e += wstride) {
        long long s, d; edge_nodes(ei, Eg, e, s, d);
        float alpha = 0.f;
        if (lane < H) {
            float v  = lrelu(a_s[s * H + lane] + a_d[d * H + lane]);
            float ex = __expf(v - fdec(m[d * H + lane]));
            alpha = ex * __builtin_amdgcn_rcpf(denom[d * H + lane] + EPS_F);
        }
        for (int ch = lane; ch < F; ch += 32) {
            int   hd = ch / Cc;
            float al = __shfl(alpha, hd, 32);    // broadcast per-head alpha across wave
            atomicAdd(&out[d * (long long)F + ch], hbuf[s * (long long)F + ch] * al);
        }
    }
}

// =====================================================================================

extern "C" void kernel_launch(void* const* d_in, const int* in_sizes, int n_in,
                              void* d_out, int out_size, void* d_ws, size_t ws_size,
                              hipStream_t stream) {
    const float*     inp      = (const float*)d_in[0];
    const long long* ei       = (const long long*)d_in[1];   // [2, E] int64
    const float*     W1       = (const float*)d_in[2];       // [128,4,32]
    const float*     att_src1 = (const float*)d_in[3];       // [4,32]
    const float*     att_dst1 = (const float*)d_in[4];
    const float*     bias1    = (const float*)d_in[5];       // [128]
    const float*     W2       = (const float*)d_in[6];       // [128,1,64]
    const float*     att_src2 = (const float*)d_in[7];       // [1,64]
    const float*     att_dst2 = (const float*)d_in[8];
    const float*     bias2    = (const float*)d_in[9];       // [64]

    const int  F_IN = 128, HEADS = 4, HID = 32, NCLS = 64, D2 = HEADS * HID; // 128
    const long long N    = in_sizes[0] / F_IN;
    const long long E    = in_sizes[1] / 2;
    const long long Etot = E + N;

    // -------- workspace carve-out (256B aligned) --------
    char*  ws  = (char*)d_ws;
    size_t off = 0;
    auto carve = [&](size_t bytes) -> void* {
        off = (off + 255) & ~(size_t)255;
        void* p = ws + off;
        off += bytes;
        return p;
    };
    float*        h_f32   = (float*)       carve((size_t)N * 128 * sizeof(float)); // h1 then h2
    float*        agg     = (float*)       carve((size_t)N * 128 * sizeof(float)); // x1
    f16*          f16buf  = (f16*)         carve((size_t)N * 128 * sizeof(f16));   // A operand
    f16*          Wt      = (f16*)         carve((size_t)128 * 128 * sizeof(f16));
    float*        a_s     = (float*)       carve((size_t)N * HEADS * sizeof(float));
    float*        a_d     = (float*)       carve((size_t)N * HEADS * sizeof(float));
    unsigned int* mmax    = (unsigned int*)carve((size_t)N * HEADS * sizeof(unsigned int));
    float*        denom   = (float*)       carve((size_t)N * HEADS * sizeof(float));
    (void)ws_size; (void)n_in; (void)out_size;

    const int  B = 256;
    auto cdiv = [](long long a, long long b) { return (int)((a + b - 1) / b); };
    const long long Mtiles = (N + 15) / 16;
    const int gemm_grid = cdiv(Mtiles, B / 32);

    // ================= Layer 1 =================
    k_f32_to_f16<<<cdiv(N * 128, B), B, 0, stream>>>(inp, f16buf, N * 128);
    k_transpose_to_f16<<<cdiv(128 * D2, B), B, 0, stream>>>(W1, Wt, 128, D2);
    k_gemm_wmma<8><<<gemm_grid, B, 0, stream>>>(f16buf, Wt, h_f32, (int)N, (int)Mtiles);
    k_node_att<<<cdiv(N * HEADS, B), B, 0, stream>>>(h_f32, att_src1, att_dst1, a_s, a_d,
                                                     (int)N, HEADS, HID);
    k_init_softmax<<<cdiv(N * HEADS, B), B, 0, stream>>>(mmax, denom, N * HEADS);
    k_edge_max<<<cdiv(Etot, B), B, 0, stream>>>(ei, E, Etot, a_s, a_d, mmax, HEADS);
    k_edge_expsum<<<cdiv(Etot, B), B, 0, stream>>>(ei, E, Etot, a_s, a_d, mmax, denom, HEADS);
    k_init_bias<<<cdiv(N * D2, B), B, 0, stream>>>(agg, bias1, N, D2);
    k_edge_aggregate<<<cdiv(Etot, B / 32), B, 0, stream>>>(ei, E, Etot, a_s, a_d, mmax,
                                                           denom, h_f32, agg, HEADS, HID);

    // ================= Layer 2 =================
    k_f32_to_f16<<<cdiv(N * 128, B), B, 0, stream>>>(agg, f16buf, N * 128);
    k_transpose_to_f16<<<cdiv(128 * NCLS, B), B, 0, stream>>>(W2, Wt, 128, NCLS);
    k_gemm_wmma<4><<<gemm_grid, B, 0, stream>>>(f16buf, Wt, h_f32, (int)N, (int)Mtiles);
    k_node_att<<<cdiv(N * 1, B), B, 0, stream>>>(h_f32, att_src2, att_dst2, a_s, a_d,
                                                 (int)N, 1, NCLS);
    k_init_softmax<<<cdiv(N, B), B, 0, stream>>>(mmax, denom, N);
    k_edge_max<<<cdiv(Etot, B), B, 0, stream>>>(ei, E, Etot, a_s, a_d, mmax, 1);
    k_edge_expsum<<<cdiv(Etot, B), B, 0, stream>>>(ei, E, Etot, a_s, a_d, mmax, denom, 1);
    k_init_bias<<<cdiv(N * NCLS, B), B, 0, stream>>>((float*)d_out, bias2, N, NCLS);
    k_edge_aggregate<<<cdiv(Etot, B / 32), B, 0, stream>>>(ei, E, Etot, a_s, a_d, mmax,
                                                           denom, h_f32, (float*)d_out,
                                                           1, NCLS);
}
